// ConditionalRNN_420906795196
// MI455X (gfx1250) — matrix operand order repaired and checked by reading, not compiled, not run on hardware
//
#include <hip/hip_runtime.h>

typedef __attribute__((ext_vector_type(16))) __bf16        v16bf;
typedef __attribute__((ext_vector_type(8)))  float         v8f;
typedef __attribute__((ext_vector_type(8)))  unsigned int  v8u;
typedef int v4i_ __attribute__((__vector_size__(4 * sizeof(int))));
typedef __attribute__((address_space(3))) v4i_ lds_v4i_;

#define B_   256
#define T_   2048
#define D_   64
#define U_   128
#define CND_ 32
#define G_   512   // 4*U

#if defined(__AMDGCN__) && __has_builtin(__builtin_amdgcn_global_load_async_to_lds_b128)
#define USE_ASYNC_LDS 1
#else
#define USE_ASYNC_LDS 0
#endif

union PackBF { __bf16 h[2]; unsigned int u; };

__device__ __forceinline__ float sigmoidf_(float x) {
    return 1.0f / (1.0f + __expf(-x));
}

__device__ __forceinline__ void wait_async0_() {
#if USE_ASYNC_LDS
#if __has_builtin(__builtin_amdgcn_s_wait_asynccnt)
    __builtin_amdgcn_s_wait_asynccnt(0);
#else
    asm volatile("s_wait_asynccnt 0x0" ::: "memory");
#endif
#endif
}

// One workgroup per 16 batch rows; persistent across all T steps.
// 8 waves; wave w owns column tiles nt = 4w..4w+3 (each 16 of the 512 gate cols).
__global__ __launch_bounds__(256) void lstm_scan_kernel(
    const float* __restrict__ x,     // [B, T, D]
    const float* __restrict__ cond,  // [B, CND]
    const float* __restrict__ Wc,    // [CND, U]
    const float* __restrict__ bc,    // [U]
    const float* __restrict__ Wk,    // [D, 4U]
    const float* __restrict__ Uk,    // [U, 4U]
    const float* __restrict__ bias,  // [4U]
    float* __restrict__ out)         // [B, U]
{
    __shared__ float        xbuf[2][16 * D_];    // double-buffered x tile  (8 KB)
    __shared__ float        gbuf[16 * G_];       // pre-activation gates    (32 KB)
    __shared__ unsigned int hA[4 * 32 * 8];      // h in bf16 A-layout      (4 KB)

    const int tid    = threadIdx.x;
    const int lane   = tid & 31;
    const int wave   = tid >> 5;
    const int b0     = blockIdx.x * 16;
    const int mrow   = lane & 15;
    const int hiHalf = lane >> 4;

    // per-thread 16B slice of the x tile
    const int xr = tid >> 4;
    const int xc = (tid & 15) * 4;
    const float* xsrc = x + (size_t)(b0 + xr) * T_ * D_ + xc;   // + t*D_ per step

    // ---- kick off staging of x tile for t=0 ----
#if USE_ASYNC_LDS
    __builtin_amdgcn_global_load_async_to_lds_b128(
        (v4i_*)xsrc,
        (lds_v4i_*)(void*)&xbuf[0][xr * D_ + xc],
        0, 0);
#else
    float4 xv = *(const float4*)xsrc;
#endif

    // ---- Load Wk and Uk into registers, bf16, B-matrix layout ----
    // B 32x16 layout: K for VGPR v: base + hiHalf*8 + (v<4 ? 2v : 2v+8)
    v16bf wk[4][2];   // [nt4][kb over D=64]
    v16bf uk[4][4];   // [nt4][kb over U=128]
    float bv[4];
#pragma unroll
    for (int nt4 = 0; nt4 < 4; ++nt4) {
        const int nt = wave * 4 + nt4;
        const int n  = nt * 16 + mrow;
        bv[nt4] = bias[n];
#pragma unroll
        for (int kb = 0; kb < 2; ++kb) {
#pragma unroll
            for (int v = 0; v < 8; ++v) {
                const int k = kb * 32 + hiHalf * 8 + ((v < 4) ? 2 * v : 2 * v + 8);
                wk[nt4][kb][2 * v]     = (__bf16)Wk[(k    ) * G_ + n];
                wk[nt4][kb][2 * v + 1] = (__bf16)Wk[(k + 1) * G_ + n];
            }
        }
#pragma unroll
        for (int kb = 0; kb < 4; ++kb) {
#pragma unroll
            for (int v = 0; v < 8; ++v) {
                const int k = kb * 32 + hiHalf * 8 + ((v < 4) ? 2 * v : 2 * v + 8);
                uk[nt4][kb][2 * v]     = (__bf16)Uk[(k    ) * G_ + n];
                uk[nt4][kb][2 * v + 1] = (__bf16)Uk[(k + 1) * G_ + n];
            }
        }
    }

    // ---- h0 = c0 = cond @ Wc + bc (tiny: 32-long dots on VALU) ----
    const int em = tid & 15;
    const int u0 = (tid >> 4) * 8;
    float cst[8], hlast[8];
#pragma unroll
    for (int s = 0; s < 8; ++s) {
        const int u = u0 + s;
        float acc = bc[u];
        for (int k = 0; k < CND_; ++k)
            acc += cond[(b0 + em) * CND_ + k] * Wc[k * U_ + u];
        cst[s]   = acc;
        hlast[s] = acc;
    }
    // Write h0 to hA in bf16 A-matrix layout.
    {
        const int kb  = u0 >> 5;
        const int ku0 = u0 & 31;
        const int lnA = (ku0 & 8) ? (16 + em) : em;
        const int v0  = (ku0 & 16) ? 4 : 0;
#pragma unroll
        for (int s2 = 0; s2 < 4; ++s2) {
            PackBF p;
            p.h[0] = (__bf16)hlast[2 * s2];
            p.h[1] = (__bf16)hlast[2 * s2 + 1];
            hA[(kb * 32 + lnA) * 8 + v0 + s2] = p.u;
        }
    }
    __syncthreads();

    // ============================ scan over T ============================
    for (int t = 0; t < T_; ++t) {
#if USE_ASYNC_LDS
        wait_async0_();                 // x tile (issued at t-1) landed in LDS
#else
        *(float4*)(&xbuf[t & 1][xr * D_ + xc]) = xv;   // stage prefetched regs
#endif
        __syncthreads();   // sync A: xbuf[t&1] + hA visible to all waves

        // pull the stream HBM->L2 well ahead
        if (t + 8 < T_) __builtin_prefetch(xsrc + (size_t)(t + 8) * D_, 0, 1);

        // issue next step's staging so it overlaps this step's compute
        if (t + 1 < T_) {
#if USE_ASYNC_LDS
            __builtin_amdgcn_global_load_async_to_lds_b128(
                (v4i_*)(xsrc + (size_t)(t + 1) * D_),
                (lds_v4i_*)(void*)&xbuf[(t + 1) & 1][xr * D_ + xc],
                0, 0);
#else
            xv = *(const float4*)(xsrc + (size_t)(t + 1) * D_);
#endif
        }

        const float* xb = xbuf[t & 1];

        // ---- build A fragments ----
        // A-layout K for VGPR v: base + hiHalf*8 + (v<4 ? 2v : 2v+8)
        v16bf ax[2];
#pragma unroll
        for (int kb = 0; kb < 2; ++kb) {
#pragma unroll
            for (int v = 0; v < 8; ++v) {
                const int k = kb * 32 + hiHalf * 8 + ((v < 4) ? 2 * v : 2 * v + 8);
                ax[kb][2 * v]     = (__bf16)xb[mrow * D_ + k];
                ax[kb][2 * v + 1] = (__bf16)xb[mrow * D_ + k + 1];
            }
        }
        v16bf ah[4];
#pragma unroll
        for (int kb = 0; kb < 4; ++kb) {
            v8u hu;
#pragma unroll
            for (int v = 0; v < 8; ++v)
                hu[v] = hA[(kb * 32 + lane) * 8 + v];
            ah[kb] = __builtin_bit_cast(v16bf, hu);
        }

        // ---- WMMA chain: g = b + x_t@Wk + h@Uk, per 16x16 tile ----
#pragma unroll
        for (int nt4 = 0; nt4 < 4; ++nt4) {
            const int nt = wave * 4 + nt4;
            v8f cc;
#pragma unroll
            for (int v = 0; v < 8; ++v) cc[v] = bv[nt4];
#pragma unroll
            for (int kb = 0; kb < 2; ++kb)
                cc = __builtin_amdgcn_wmma_f32_16x16x32_bf16(
                        false, ax[kb], false, wk[nt4][kb], (short)0, cc, false, false);
#pragma unroll
            for (int kb = 0; kb < 4; ++kb)
                cc = __builtin_amdgcn_wmma_f32_16x16x32_bf16(
                        false, ah[kb], false, uk[nt4][kb], (short)0, cc, false, false);
            // D layout: VGPR v -> row hiHalf*8+v, col mrow
#pragma unroll
            for (int v = 0; v < 8; ++v)
                gbuf[(hiHalf * 8 + v) * G_ + nt * 16 + mrow] = cc[v];
        }
        __syncthreads();   // sync B: gates ready; safe to overwrite hA below

        // ---- elementwise gate update (c stays in registers) ----
        float hnew[8];
#pragma unroll
        for (int s = 0; s < 8; ++s) {
            const int u = u0 + s;
            const float ig = sigmoidf_(gbuf[em * G_ + u]);
            const float fg = sigmoidf_(gbuf[em * G_ + U_ + u]);
            const float cb = tanhf(     gbuf[em * G_ + 2 * U_ + u]);
            const float og = sigmoidf_(gbuf[em * G_ + 3 * U_ + u]);
            const float cn = fg * cst[s] + ig * cb;
            cst[s]   = cn;
            hnew[s]  = og * tanhf(cn);
            hlast[s] = hnew[s];
        }
        // write h_new to hA (bf16 A-layout) for next step
        {
            const int kb  = u0 >> 5;
            const int ku0 = u0 & 31;
            const int lnA = (ku0 & 8) ? (16 + em) : em;
            const int v0  = (ku0 & 16) ? 4 : 0;
#pragma unroll
            for (int s2 = 0; s2 < 4; ++s2) {
                PackBF p;
                p.h[0] = (__bf16)hnew[2 * s2];
                p.h[1] = (__bf16)hnew[2 * s2 + 1];
                hA[(kb * 32 + lnA) * 8 + v0 + s2] = p.u;
            }
        }
        // loop-top barrier orders hA writes vs. next step's reads
    }

    // ---- write last hidden state ----
#pragma unroll
    for (int s = 0; s < 8; ++s)
        out[(b0 + em) * U_ + u0 + s] = hlast[s];
}

extern "C" void kernel_launch(void* const* d_in, const int* in_sizes, int n_in,
                              void* d_out, int out_size, void* d_ws, size_t ws_size,
                              hipStream_t stream) {
    (void)in_sizes; (void)n_in; (void)d_ws; (void)ws_size; (void)out_size;
    const float* x    = (const float*)d_in[0];  // [B,T,D]
    const float* cond = (const float*)d_in[1];  // [B,COND]
    const float* Wc   = (const float*)d_in[2];  // [COND,U]
    const float* bc   = (const float*)d_in[3];  // [U]
    const float* Wk   = (const float*)d_in[4];  // [D,4U]
    const float* Uk   = (const float*)d_in[5];  // [U,4U]
    const float* b    = (const float*)d_in[6];  // [4U]
    float* out = (float*)d_out;                 // [B,U]

    lstm_scan_kernel<<<dim3(B_ / 16), dim3(256), 0, stream>>>(
        x, cond, Wc, bc, Wk, Uk, b, out);
}
